// QuantAttentionFused_68427418960041
// MI455X (gfx1250) — matrix operand order repaired and checked by reading, not compiled, not run on hardware
//
#include <hip/hip_runtime.h>
#include <hip/hip_bf16.h>
#include <math.h>

typedef __attribute__((ext_vector_type(16))) __bf16 bf16x16;
typedef __attribute__((ext_vector_type(8)))  __bf16 bf16x8;
typedef __attribute__((ext_vector_type(4)))  __bf16 bf16x4;
typedef __attribute__((ext_vector_type(8)))  float  f32x8;
typedef __attribute__((ext_vector_type(4)))  int    i32x4;

#define SEQ     2048
#define HID     4096
#define QKV_OUT 12288
#define HEADS   32
#define HDIM    128
#define GROUPSZ 128

static __device__ __forceinline__ bf16x16 lds_load_frag16(const __bf16* p0, const __bf16* p1) {
  bf16x16 r;
  *(bf16x8*)&r       = *(const bf16x8*)p0;
  *((bf16x8*)&r + 1) = *(const bf16x8*)p1;
  return r;
}

// ---------------------------------------------------------------- fp32 -> bf16
__global__ __launch_bounds__(256)
void f32_to_bf16_kernel(const float* __restrict__ in, __bf16* __restrict__ out, int n) {
  int i = blockIdx.x * blockDim.x + threadIdx.x;
  int stride = gridDim.x * blockDim.x;
  for (; i < n; i += stride) out[i] = (__bf16)in[i];
}

// ------------------------------------------- AWQ dequant + GEMM via bf16 WMMA
// X: [M,K] bf16   Qw: [K,N] int32(0..15)   Sc/Zp: [K/128, N] f32
// Block tile 128x128, 256 threads = 8 waves, wave tile 32x64 (2x4 WMMA 16x16).
__global__ __launch_bounds__(256)
void dequant_gemm_wmma(const __bf16* __restrict__ X,
                       const int*   __restrict__ Qw,
                       const float* __restrict__ Sc,
                       const float* __restrict__ Zp,
                       float*  __restrict__ outF,   // fp32 output (or null)
                       __bf16* __restrict__ outB,   // bf16 output (or null)
                       int M, int N, int K)
{
  __shared__ __align__(16) __bf16 s_a[128][40];   // A tile 128x32 (+8 pad)
  __shared__ __align__(16) __bf16 s_b[128][40];   // B tile transposed: [n][k]
  __shared__ __align__(16) float s_scale[128];
  __shared__ __align__(16) float s_mzs[128];      // -zero*scale per column

  const int tid  = threadIdx.x;
  const int lane = tid & 31;
  const int wid  = tid >> 5;
  const int lm   = lane & 15;
  const int lh   = lane >> 4;
  const int wr   = wid >> 1;          // 0..3 (M dir, 32 rows each)
  const int wc   = wid & 1;           // 0..1 (N dir, 64 cols each)
  const int m0   = blockIdx.y * 128;
  const int n0   = blockIdx.x * 128;

  f32x8 acc[2][4];
  #pragma unroll
  for (int a = 0; a < 2; ++a)
    #pragma unroll
    for (int b = 0; b < 4; ++b) acc[a][b] = f32x8{};

  // A-tile staging: two 16B chunks per thread, row pointers hoisted
  const int arow = tid >> 2;          // 0..63 (and +64)
  const int acol = (tid & 3) * 8;     // 0,8,16,24
  const __bf16* xrow = X + (size_t)(m0 + arow) * K + acol;

  // B-dequant micro-tile: 4 columns x 4 k-rows per thread
  const int tn = (lane) * 4;          // 0..124 within block's 128 columns
  const int tk = wid * 4;             // 0..28 within 32-k tile
  const int*  qbase = Qw + n0 + tn;   // column base, rows added per step

  const int nGroups = K / GROUPSZ;
  for (int g = 0; g < nGroups; ++g) {
    // per-group scale and (-zero*scale) for this block's 128 columns
    if (tid < 128) {
      float sc = Sc[(size_t)g * N + n0 + tid];
      float zp = Zp[(size_t)g * N + n0 + tid];
      s_scale[tid] = sc;
      s_mzs[tid]   = -zp * sc;
    }
    #pragma unroll 1
    for (int t = 0; t < GROUPSZ / 32; ++t) {
      const int kk = g * GROUPSZ + t * 32;
      __syncthreads();

      // A tile: 128x32 bf16, coalesced b128 chunks
      *(bf16x8*)&s_a[arow][acol]      = *(const bf16x8*)(xrow + kk);
      *(bf16x8*)&s_a[arow + 64][acol] = *(const bf16x8*)(xrow + (size_t)64 * K + kk);

      // B tile: b128 weight loads (4 cols), fma dequant, b64 transposed stores
      {
        const int* qp = qbase + (size_t)(kk + tk) * N;
        i32x4 qv[4];
        #pragma unroll
        for (int dk = 0; dk < 4; ++dk)
          qv[dk] = *(const i32x4*)(qp + (size_t)dk * N);
        if (kk + 32 < K)   // prefetch next K-tile of the weight stream
          __builtin_prefetch(qp + (size_t)32 * N, 0, 1);
        float sc4[4], mz4[4];
        #pragma unroll
        for (int j = 0; j < 4; ++j) {
          sc4[j] = s_scale[tn + j];
          mz4[j] = s_mzs[tn + j];
        }
        #pragma unroll
        for (int j = 0; j < 4; ++j) {
          bf16x4 col;
          #pragma unroll
          for (int dk = 0; dk < 4; ++dk)
            col[dk] = (__bf16)fmaf((float)qv[dk][j], sc4[j], mz4[j]);
          *(bf16x4*)&s_b[tn + j][tk] = col;   // k-contiguous: one ds_store_b64
        }
      }
      __syncthreads();

      bf16x16 afrag[2], bfrag[4];
      #pragma unroll
      for (int sm = 0; sm < 2; ++sm) {
        const __bf16* pa = &s_a[wr * 32 + sm * 16 + lm][lh * 8];
        afrag[sm] = lds_load_frag16(pa, pa + 16);     // K 0-7/16-23 | 8-15/24-31
      }
      #pragma unroll
      for (int sn = 0; sn < 4; ++sn) {
        const __bf16* pb = &s_b[wc * 64 + sn * 16 + lm][lh * 8];
        bfrag[sn] = lds_load_frag16(pb, pb + 16);     // K 0-7/16-23 | 8-15/24-31
      }
      #pragma unroll
      for (int sm = 0; sm < 2; ++sm)
        #pragma unroll
        for (int sn = 0; sn < 4; ++sn)
          acc[sm][sn] = __builtin_amdgcn_wmma_f32_16x16x32_bf16(
              false, afrag[sm], false, bfrag[sn], (short)0, acc[sm][sn], false, false);
    }
  }

  // C layout: VGPR r -> row r (lanes 0-15) / r+8 (lanes 16-31), col = lane&15
  const int rowBase = m0 + wr * 32;
  const int colBase = n0 + wc * 64;
  if (outF) {
    #pragma unroll
    for (int sm = 0; sm < 2; ++sm)
      #pragma unroll
      for (int sn = 0; sn < 4; ++sn)
        #pragma unroll
        for (int r = 0; r < 8; ++r)
          outF[(size_t)(rowBase + sm * 16 + r + 8 * lh) * N +
               colBase + sn * 16 + lm] = acc[sm][sn][r];
  } else {
    #pragma unroll
    for (int sm = 0; sm < 2; ++sm)
      #pragma unroll
      for (int sn = 0; sn < 4; ++sn)
        #pragma unroll
        for (int r = 0; r < 8; ++r)
          outB[(size_t)(rowBase + sm * 16 + r + 8 * lh) * N +
               colBase + sn * 16 + lm] = (__bf16)acc[sm][sn][r];
  }
}

// ------------------------------------------------- RoPE + split to [h][s][d]
__global__ __launch_bounds__(128)
void rope_split_kernel(const __bf16* __restrict__ proj,
                       __bf16* __restrict__ Qo,
                       __bf16* __restrict__ Ko,
                       __bf16* __restrict__ Vo)
{
  const int s = blockIdx.x;
  const int d = threadIdx.x;          // 0..127
  const int i = d & 63;
  float inv = __powf(10000.0f, -(float)i / 64.0f);
  float ang = (float)s * inv;
  float c = __cosf(ang), sn = __sinf(ang);
  const float qscale = 0.08838834764831845f;  // 1/sqrt(128), folded into Q
  const int   dp     = (d + 64) & 127;
  const float rsign  = (d < 64) ? -1.0f : 1.0f;
  const __bf16* base = proj + (size_t)s * QKV_OUT;
  for (int h = 0; h < HEADS; ++h) {
    float x, xr;
    x  = (float)base[0 * HID + h * HDIM + d];
    xr = rsign * (float)base[0 * HID + h * HDIM + dp];
    Qo[((size_t)h * SEQ + s) * HDIM + d] = (__bf16)((x * c + xr * sn) * qscale);
    x  = (float)base[1 * HID + h * HDIM + d];
    xr = rsign * (float)base[1 * HID + h * HDIM + dp];
    Ko[((size_t)h * SEQ + s) * HDIM + d] = (__bf16)(x * c + xr * sn);
    Vo[((size_t)h * SEQ + s) * HDIM + d] = base[2 * HID + h * HDIM + d];
  }
}

// ------------------------------------------- causal flash attention via WMMA
// 4 waves/block; block = 64 q-rows of one head; KV tiles of 32.
__global__ __launch_bounds__(128)
void flash_attn_wmma(const __bf16* __restrict__ Q,
                     const __bf16* __restrict__ Km,
                     const __bf16* __restrict__ V,
                     __bf16* __restrict__ attn)
{
  __shared__ __align__(16) __bf16 s_q[64][136];   // Q tile row-major (+8 pad)
  __shared__ __align__(16) __bf16 s_k[32][136];   // K tile row-major
  __shared__ __align__(16) __bf16 s_vt[128][40];  // V tile transposed [d][kv]
  __shared__ __align__(16) __bf16 s_p[4][16][32]; // per-wave P (C->A relayout)

  const int tid  = threadIdx.x;
  const int lane = tid & 31;
  const int wid  = tid >> 5;          // 0..3
  const int lm   = lane & 15;
  const int lh   = lane >> 4;
  const int q0   = blockIdx.x * 64;
  const int head = blockIdx.y;
  const __bf16* qh = Q  + (size_t)head * SEQ * HDIM;
  const __bf16* kh = Km + (size_t)head * SEQ * HDIM;
  const __bf16* vh = V  + (size_t)head * SEQ * HDIM;

  for (int idx = tid; idx < 64 * 16; idx += 128) {
    int r = idx >> 4, c = (idx & 15) * 8;
    *(bf16x8*)&s_q[r][c] = *(const bf16x8*)&qh[(size_t)(q0 + r) * HDIM + c];
  }

  f32x8 o[8];
  #pragma unroll
  for (int t = 0; t < 8; ++t) o[t] = f32x8{};
  float rowM[8], rowL[8];
  #pragma unroll
  for (int r = 0; r < 8; ++r) { rowM[r] = -3.0e38f; rowL[r] = 0.0f; }

  const int jEnd = (q0 >> 5) + 2;     // causal bound for this q block
  for (int j = 0; j < jEnd; ++j) {
    __syncthreads();
    for (int idx = tid; idx < 32 * 16; idx += 128) {
      int r = idx >> 4, c = (idx & 15) * 8;
      *(bf16x8*)&s_k[r][c] = *(const bf16x8*)&kh[(size_t)(j * 32 + r) * HDIM + c];
    }
    {
      int d = tid;
      #pragma unroll 4
      for (int kv = 0; kv < 32; ++kv)
        s_vt[d][kv] = vh[(size_t)(j * 32 + kv) * HDIM + d];
    }
    __syncthreads();

    // S = Q K^T : this wave's 16 rows x 32 kv cols, reduce over d=128
    f32x8 s_acc[2];
    s_acc[0] = f32x8{}; s_acc[1] = f32x8{};
    #pragma unroll
    for (int kt = 0; kt < 4; ++kt) {
      int db = kt * 32;
      const __bf16* pa = &s_q[wid * 16 + lm][db + lh * 8];
      bf16x16 af = lds_load_frag16(pa, pa + 16);
      #pragma unroll
      for (int sn = 0; sn < 2; ++sn) {
        const __bf16* pb = &s_k[sn * 16 + lm][db + lh * 8];
        bf16x16 bf_ = lds_load_frag16(pb, pb + 16);
        s_acc[sn] = __builtin_amdgcn_wmma_f32_16x16x32_bf16(
            false, af, false, bf_, (short)0, s_acc[sn], false, false);
      }
    }

    // causal mask + online softmax (row r+8*lh owned across 16-lane half)
    #pragma unroll
    for (int r = 0; r < 8; ++r) {
      int qrow = q0 + wid * 16 + r + 8 * lh;
      float s0 = s_acc[0][r]; if (j * 32 + lm      > qrow) s0 = -3.0e38f;
      float s1 = s_acc[1][r]; if (j * 32 + 16 + lm > qrow) s1 = -3.0e38f;
      float m = fmaxf(s0, s1);
      m = fmaxf(m, __shfl_xor(m, 1));
      m = fmaxf(m, __shfl_xor(m, 2));
      m = fmaxf(m, __shfl_xor(m, 4));
      m = fmaxf(m, __shfl_xor(m, 8));
      float Mnew = fmaxf(rowM[r], m);
      float f  = __expf(rowM[r] - Mnew);
      float p0 = __expf(s0 - Mnew);
      float p1 = __expf(s1 - Mnew);
      float l = p0 + p1;
      l += __shfl_xor(l, 1);
      l += __shfl_xor(l, 2);
      l += __shfl_xor(l, 4);
      l += __shfl_xor(l, 8);
      rowL[r] = rowL[r] * f + l;
      rowM[r] = Mnew;
      #pragma unroll
      for (int t = 0; t < 8; ++t) o[t][r] *= f;
      s_p[wid][r + 8 * lh][lm]      = (__bf16)p0;
      s_p[wid][r + 8 * lh][16 + lm] = (__bf16)p1;
    }
    __syncthreads();

    // O += P V  (P 16x32 as A-frag, V^T columns as B-frags)
    const __bf16* pp = &s_p[wid][lm][lh * 8];
    bf16x16 pf = lds_load_frag16(pp, pp + 16);
    #pragma unroll
    for (int t = 0; t < 8; ++t) {
      const __bf16* pb = &s_vt[t * 16 + lm][lh * 8];
      bf16x16 vf = lds_load_frag16(pb, pb + 16);
      o[t] = __builtin_amdgcn_wmma_f32_16x16x32_bf16(
          false, pf, false, vf, (short)0, o[t], false, false);
    }
  }

  #pragma unroll
  for (int r = 0; r < 8; ++r) {
    float invL = 1.0f / rowL[r];
    int row = q0 + wid * 16 + r + 8 * lh;
    #pragma unroll
    for (int t = 0; t < 8; ++t)
      attn[(size_t)row * HID + head * HDIM + t * 16 + lm] =
          (__bf16)(o[t][r] * invL);
  }
}

// ---------------------------------------------------------------------- launch
extern "C" void kernel_launch(void* const* d_in, const int* in_sizes, int n_in,
                              void* d_out, int out_size, void* d_ws, size_t ws_size,
                              hipStream_t stream) {
  (void)in_sizes; (void)n_in; (void)out_size; (void)ws_size;
  const float* hidden = (const float*)d_in[0];
  // d_in[1] = position_ids (arange, implicit)
  const int*   qkv_qw = (const int*)  d_in[2];
  const float* qkv_sc = (const float*)d_in[3];
  const float* qkv_zp = (const float*)d_in[4];
  const int*   o_qw   = (const int*)  d_in[5];
  const float* o_sc   = (const float*)d_in[6];
  const float* o_zp   = (const float*)d_in[7];
  float* out = (float*)d_out;

  char* ws = (char*)d_ws;
  __bf16* Xbf   = (__bf16*)ws; ws += (size_t)SEQ * HID * 2;       // 16 MiB
  __bf16* projb = (__bf16*)ws; ws += (size_t)SEQ * QKV_OUT * 2;   // 48 MiB
  __bf16* Qb    = (__bf16*)ws; ws += (size_t)HEADS * SEQ * HDIM * 2;
  __bf16* Kb    = (__bf16*)ws; ws += (size_t)HEADS * SEQ * HDIM * 2;
  __bf16* Vb    = (__bf16*)ws; ws += (size_t)HEADS * SEQ * HDIM * 2;
  __bf16* Ab    = (__bf16*)ws; ws += (size_t)SEQ * HID * 2;       // total 128 MiB

  f32_to_bf16_kernel<<<1024, 256, 0, stream>>>(hidden, Xbf, SEQ * HID);
  dequant_gemm_wmma<<<dim3(QKV_OUT / 128, SEQ / 128), 256, 0, stream>>>(
      Xbf, qkv_qw, qkv_sc, qkv_zp, nullptr, projb, SEQ, QKV_OUT, HID);
  rope_split_kernel<<<SEQ, 128, 0, stream>>>(projb, Qb, Kb, Vb);
  flash_attn_wmma<<<dim3(SEQ / 64, HEADS), 128, 0, stream>>>(Qb, Kb, Vb, Ab);
  dequant_gemm_wmma<<<dim3(HID / 128, SEQ / 128), 256, 0, stream>>>(
      Ab, o_qw, o_sc, o_zp, out, nullptr, SEQ, HID, HID);
}
